// CrossAttention_39702677684942
// MI455X (gfx1250) — compile-verified, matrix operand order checked by reading, not run on hardware
//
#include <hip/hip_runtime.h>
#include <hip/hip_bf16.h>

typedef __bf16 bf16_t;
typedef __attribute__((ext_vector_type(16))) __bf16 bf16x16;
typedef __attribute__((ext_vector_type(8)))  __bf16 bf16x8_t;
typedef __attribute__((ext_vector_type(8)))  float  f32x8;

#define B_ 8
#define N_ 8192
#define C_ 256
#define H_ 4
#define M_ (B_ * N_)        // 65536 rows
#define NCHUNK_ 8
#define CTOK_ (N_ / NCHUNK_)  // 1024 tokens per attention chunk

// CDNA5 16-bit A/B fragment (ISA 7.12.2): element j, lane half = lane>>4:
//   K(j, half) = (j<8 ? j : j+8) + 8*half
// => fragment = two contiguous 8-element (16B) runs at K = 8*half and K = 16+8*half.
__device__ __forceinline__ bf16x16 load_frag16(const bf16_t* base /* = row + 8*half */) {
    bf16x8_t lo = *(const bf16x8_t*)(base);        // K: 8*half .. 8*half+7
    bf16x8_t hi = *(const bf16x8_t*)(base + 16);   // K: 16+8*half .. +7
    return __builtin_shufflevector(lo, hi, 0, 1, 2, 3, 4, 5, 6, 7,
                                   8, 9, 10, 11, 12, 13, 14, 15);
}

// gfx1250 async global->LDS copy, 16B per lane. LDS dest = low 32 bits of the
// generic shared-memory address (LDS aperture keeps the byte offset in [31:0]).
__device__ __forceinline__ void async_load_b128(const void* lds_dst, const void* gsrc) {
    unsigned int       ldsoff = (unsigned int)(unsigned long long)lds_dst;
    unsigned long long gaddr  = (unsigned long long)gsrc;
    asm volatile("global_load_async_to_lds_b128 %0, %1, off"
                 :: "v"(ldsoff), "v"(gaddr) : "memory");
}

// ---------------------------------------------------------------------------
// Cast fp32 -> bf16 (weights)
// ---------------------------------------------------------------------------
__global__ void cast_f32_to_bf16(const float* __restrict__ in,
                                 bf16_t* __restrict__ out, int n) {
    int i = blockIdx.x * blockDim.x + threadIdx.x;
    if (i < n) out[i] = (bf16_t)in[i];
}

// ---------------------------------------------------------------------------
// Y = bf16( X_f32 @ W_bf16^T ), X:[M_,256] fp32, W:[256,256] bf16 row-major
// block = 256 threads = 8 waves; wave tile 16(M) x 32(N): 1 A frag, 2 B frags,
// 2 WMMA per k-step. Block tile 64 x 64. grid (M_/64, C_/64)
// ---------------------------------------------------------------------------
__global__ void proj_gemm_kernel(const float* __restrict__ X,
                                 const bf16_t* __restrict__ W,
                                 bf16_t* __restrict__ Y) {
    const int tid  = threadIdx.x;
    const int wave = tid >> 5, lane = tid & 31;
    const int half = lane >> 4, l16 = lane & 15;
    const int rowBase = blockIdx.x * 64 + (wave & 3) * 16;
    const int colBase = blockIdx.y * 64 + (wave >> 2) * 32;

    f32x8 acc0 = {}, acc1 = {};
    const float*  arow  = X + (size_t)(rowBase + l16) * C_;
    const bf16_t* brow0 = W + (size_t)(colBase + l16) * C_;
    const bf16_t* brow1 = W + (size_t)(colBase + 16 + l16) * C_;

    for (int k0 = 0; k0 < C_; k0 += 32) {
        f32x8 alo = *(const f32x8*)(arow + k0 + 8 * half);
        f32x8 ahi = *(const f32x8*)(arow + k0 + 16 + 8 * half);
        bf16x16 a;
#pragma unroll
        for (int j = 0; j < 8; ++j) {
            a[j]     = (bf16_t)alo[j];
            a[j + 8] = (bf16_t)ahi[j];
        }
        bf16x16 b0 = load_frag16(brow0 + k0 + 8 * half);
        bf16x16 b1 = load_frag16(brow1 + k0 + 8 * half);
        acc0 = __builtin_amdgcn_wmma_f32_16x16x32_bf16(false, a, false, b0,
                                                       (short)0, acc0, false, false);
        acc1 = __builtin_amdgcn_wmma_f32_16x16x32_bf16(false, a, false, b1,
                                                       (short)0, acc1, false, false);
    }
#pragma unroll
    for (int g = 0; g < 8; ++g) {
        int m = g + 8 * half;
        Y[(size_t)(rowBase + m) * C_ + colBase + l16]      = (bf16_t)acc0[g];
        Y[(size_t)(rowBase + m) * C_ + colBase + 16 + l16] = (bf16_t)acc1[g];
    }
}

// ---------------------------------------------------------------------------
// Per-channel sum of squares over tokens: part[b][chunk][c]
// ---------------------------------------------------------------------------
__global__ void colsumsq_partial(const bf16_t* __restrict__ Q,
                                 float* __restrict__ part) {
    const int b = blockIdx.x, chunk = blockIdx.y, c = threadIdx.x;
    const int TOK = N_ / 32;
    float s = 0.f;
    size_t base = ((size_t)b * N_ + (size_t)chunk * TOK) * C_ + c;
    for (int n = 0; n < TOK; ++n) {
        float v = (float)Q[base + (size_t)n * C_];
        s += v * v;
    }
    part[((size_t)b * 32 + chunk) * C_ + c] = s;
}

__global__ void finalize_inv_norm(const float* __restrict__ part,
                                  float* __restrict__ inv) {
    const int b = blockIdx.x, c = threadIdx.x;
    float s = 0.f;
    for (int ch = 0; ch < 32; ++ch) s += part[((size_t)b * 32 + ch) * C_ + c];
    float nrm = fmaxf(sqrtf(s), 1e-12f);
    inv[b * C_ + c] = 1.0f / nrm;
}

// ---------------------------------------------------------------------------
// Attention logits, N split into NCHUNK_ chunks for occupancy:
//   partS[chunk][bh][64][64] += Q_b^T K_b over the chunk's 1024 tokens.
// Double-buffered async global->LDS staging (ASYNCcnt), then transpose to a
// fragment-ready channel-major layout (row stride 40 bf16 = 80B, 16B aligned)
// so each WMMA operand is two ds_load_b128.
// grid (NCHUNK_, B_*H_), block 256 (8 waves; wave w -> tiles td=w>>1, te=2*(w&1)+{0,1})
// ---------------------------------------------------------------------------
__global__ void attn_partial_kernel(const bf16_t* __restrict__ Q,
                                    const bf16_t* __restrict__ K,
                                    float* __restrict__ partS) {
    __shared__ __align__(16) bf16_t stq[2][32][64];  // token-major async stage
    __shared__ __align__(16) bf16_t stk[2][32][64];
    __shared__ __align__(16) bf16_t Qt[64][40];      // [channel][token] + pad
    __shared__ __align__(16) bf16_t Kt[64][40];

    const int chunk = blockIdx.x, bh = blockIdx.y;
    const int b = bh >> 2, h = bh & 3;
    const int tid  = threadIdx.x;
    const int wave = tid >> 5, lane = tid & 31;
    const int half = lane >> 4, l16 = lane & 15;
    const int td  = wave >> 1;
    const int te0 = (wave & 1) * 2;

    const size_t hbase = (size_t)b * N_ * C_ + h * 64;
    const int lrow = tid >> 3;          // token within 32-token tile
    const int lcol = (tid & 7) * 8;     // channel start (16B run)
    const int tok0 = chunk * CTOK_;
    const int NT = CTOK_ / 32;          // 32 tiles

    f32x8 acc0 = {}, acc1 = {};

    // prologue: stage tile 0
    {
        const bf16_t* qs = Q + hbase + (size_t)(tok0 + lrow) * C_ + lcol;
        const bf16_t* ks = K + hbase + (size_t)(tok0 + lrow) * C_ + lcol;
        async_load_b128(&stq[0][lrow][lcol], qs);
        async_load_b128(&stk[0][lrow][lcol], ks);
    }
    int buf = 0;
    for (int i = 0; i < NT; ++i) {
        if (i + 1 < NT) {   // kick next tile, then drain current (2 ops in flight)
            const bf16_t* qs = Q + hbase + (size_t)(tok0 + (i + 1) * 32 + lrow) * C_ + lcol;
            const bf16_t* ks = K + hbase + (size_t)(tok0 + (i + 1) * 32 + lrow) * C_ + lcol;
            async_load_b128(&stq[buf ^ 1][lrow][lcol], qs);
            async_load_b128(&stk[buf ^ 1][lrow][lcol], ks);
            asm volatile("s_wait_asynccnt 0x2" ::: "memory");
        } else {
            asm volatile("s_wait_asynccnt 0x0" ::: "memory");
        }
        __syncthreads();   // staged data visible to all waves; prev frag reads done
        bf16x8_t qr = *(const bf16x8_t*)&stq[buf][lrow][lcol];
        bf16x8_t kr = *(const bf16x8_t*)&stk[buf][lrow][lcol];
#pragma unroll
        for (int j = 0; j < 8; ++j) {   // transpose into [channel][token]
            Qt[lcol + j][lrow] = qr[j];
            Kt[lcol + j][lrow] = kr[j];
        }
        __syncthreads();
        bf16x16 a  = load_frag16(&Qt[td * 16 + l16][8 * half]);
        bf16x16 b0 = load_frag16(&Kt[te0 * 16 + l16][8 * half]);
        bf16x16 b1 = load_frag16(&Kt[(te0 + 1) * 16 + l16][8 * half]);
        acc0 = __builtin_amdgcn_wmma_f32_16x16x32_bf16(false, a, false, b0,
                                                       (short)0, acc0, false, false);
        acc1 = __builtin_amdgcn_wmma_f32_16x16x32_bf16(false, a, false, b1,
                                                       (short)0, acc1, false, false);
        buf ^= 1;
    }

    float* dst = partS + ((size_t)chunk * (B_ * H_) + bh) * 4096;
#pragma unroll
    for (int g = 0; g < 8; ++g) {
        int m = g + 8 * half;
        dst[(td * 16 + m) * 64 + te0 * 16 + l16]       = acc0[g];
        dst[(td * 16 + m) * 64 + (te0 + 1) * 16 + l16] = acc1[g];
    }
}

// ---------------------------------------------------------------------------
// Sum partial S, scale by invq[d]*invk[e]*temp[h], row softmax, emit P (bf16).
// grid = B_*H_, block = 256
// ---------------------------------------------------------------------------
__global__ void attn_softmax_finalize(const float* __restrict__ partS,
                                      const float* __restrict__ invq,
                                      const float* __restrict__ invk,
                                      const float* __restrict__ temp,
                                      bf16_t* __restrict__ P) {
    __shared__ float S[4096];
    __shared__ float rinv[64];
    const int bh = blockIdx.x, b = bh >> 2, h = bh & 3;
    const int tid = threadIdx.x;

#pragma unroll
    for (int t = 0; t < 16; ++t) {
        int idx = tid + t * 256;
        float s = 0.f;
        for (int ch = 0; ch < NCHUNK_; ++ch)
            s += partS[((size_t)ch * (B_ * H_) + bh) * 4096 + idx];
        int r = idx >> 6, e = idx & 63;
        S[idx] = s * invq[b * C_ + h * 64 + r] * invk[b * C_ + h * 64 + e] * temp[h];
    }
    __syncthreads();
    if (tid < 64) {
        float mx = -3.4e38f;
        for (int e = 0; e < 64; ++e) mx = fmaxf(mx, S[tid * 64 + e]);
        float sum = 0.f;
        for (int e = 0; e < 64; ++e) {
            float ex = __expf(S[tid * 64 + e] - mx);
            S[tid * 64 + e] = ex;
            sum += ex;
        }
        rinv[tid] = 1.0f / sum;
    }
    __syncthreads();
#pragma unroll
    for (int t = 0; t < 16; ++t) {
        int idx = tid + t * 256;
        P[(size_t)bh * 4096 + idx] = (bf16_t)(S[idx] * rinv[idx >> 6]);
    }
}

// ---------------------------------------------------------------------------
// XC[b,n,64h+d] = sum_e P[b,h,d,e] * V[b,n,64h+e]
// grid (N_/64, B_*H_), block 256; wave tile 16 tokens x 32 d (2 WMMA / step)
// ---------------------------------------------------------------------------
__global__ void xcross_kernel(const bf16_t* __restrict__ V,
                              const bf16_t* __restrict__ P,
                              bf16_t* __restrict__ XC) {
    const int bh = blockIdx.y;
    const int b = bh >> 2, h = bh & 3;
    const int tid  = threadIdx.x;
    const int wave = tid >> 5, lane = tid & 31;
    const int half = lane >> 4, l16 = lane & 15;
    const int rowBase = blockIdx.x * 64 + (wave & 3) * 16;
    const int dBase   = (wave >> 2) * 32;

    f32x8 acc0 = {}, acc1 = {};
    const bf16_t* vrow  = V + ((size_t)b * N_ + rowBase + l16) * C_ + h * 64;
    const bf16_t* prow0 = P + ((size_t)bh * 64 + dBase + l16) * 64;
    const bf16_t* prow1 = P + ((size_t)bh * 64 + dBase + 16 + l16) * 64;
#pragma unroll
    for (int k0 = 0; k0 < 64; k0 += 32) {
        bf16x16 a  = load_frag16(vrow + k0 + 8 * half);
        bf16x16 b0 = load_frag16(prow0 + k0 + 8 * half);
        bf16x16 b1 = load_frag16(prow1 + k0 + 8 * half);
        acc0 = __builtin_amdgcn_wmma_f32_16x16x32_bf16(false, a, false, b0,
                                                       (short)0, acc0, false, false);
        acc1 = __builtin_amdgcn_wmma_f32_16x16x32_bf16(false, a, false, b1,
                                                       (short)0, acc1, false, false);
    }
#pragma unroll
    for (int g = 0; g < 8; ++g) {
        int m = g + 8 * half;
        size_t o = ((size_t)b * N_ + rowBase + m) * C_ + h * 64;
        XC[o + dBase + l16]      = (bf16_t)acc0[g];
        XC[o + dBase + 16 + l16] = (bf16_t)acc1[g];
    }
}

// ---------------------------------------------------------------------------
// Out = XC @ Wo^T + bo   (fp32 output); wave tile 16 x 32, grid (M_/64, C_/64)
// ---------------------------------------------------------------------------
__global__ void out_gemm_kernel(const bf16_t* __restrict__ XC,
                                const bf16_t* __restrict__ Wo,
                                const float* __restrict__ bo,
                                float* __restrict__ Out) {
    const int tid  = threadIdx.x;
    const int wave = tid >> 5, lane = tid & 31;
    const int half = lane >> 4, l16 = lane & 15;
    const int rowBase = blockIdx.x * 64 + (wave & 3) * 16;
    const int colBase = blockIdx.y * 64 + (wave >> 2) * 32;

    f32x8 acc0 = {}, acc1 = {};
    const bf16_t* arow  = XC + (size_t)(rowBase + l16) * C_;
    const bf16_t* brow0 = Wo + (size_t)(colBase + l16) * C_;
    const bf16_t* brow1 = Wo + (size_t)(colBase + 16 + l16) * C_;
    for (int k0 = 0; k0 < C_; k0 += 32) {
        bf16x16 a  = load_frag16(arow + k0 + 8 * half);
        bf16x16 b0 = load_frag16(brow0 + k0 + 8 * half);
        bf16x16 b1 = load_frag16(brow1 + k0 + 8 * half);
        acc0 = __builtin_amdgcn_wmma_f32_16x16x32_bf16(false, a, false, b0,
                                                       (short)0, acc0, false, false);
        acc1 = __builtin_amdgcn_wmma_f32_16x16x32_bf16(false, a, false, b1,
                                                       (short)0, acc1, false, false);
    }
    const float bias0 = bo[colBase + l16];
    const float bias1 = bo[colBase + 16 + l16];
#pragma unroll
    for (int g = 0; g < 8; ++g) {
        int m = g + 8 * half;
        Out[(size_t)(rowBase + m) * C_ + colBase + l16]      = acc0[g] + bias0;
        Out[(size_t)(rowBase + m) * C_ + colBase + 16 + l16] = acc1[g] + bias1;
    }
}

// ---------------------------------------------------------------------------
extern "C" void kernel_launch(void* const* d_in, const int* in_sizes, int n_in,
                              void* d_out, int out_size, void* d_ws, size_t ws_size,
                              hipStream_t stream) {
    const float* x1   = (const float*)d_in[0];
    const float* x2   = (const float*)d_in[1];
    const float* Wq   = (const float*)d_in[2];
    const float* Wk   = (const float*)d_in[3];
    const float* Wv   = (const float*)d_in[4];
    const float* Wo   = (const float*)d_in[5];
    const float* bo   = (const float*)d_in[6];
    const float* temp = (const float*)d_in[7];
    float* out = (float*)d_out;

    char* p = (char*)d_ws;
    auto alloc = [&](size_t bytes) -> void* {
        void* r = (void*)p;
        p += (bytes + 255) & ~(size_t)255;
        return r;
    };

    bf16_t* wq_bf = (bf16_t*)alloc((size_t)C_ * C_ * 2);
    bf16_t* wk_bf = (bf16_t*)alloc((size_t)C_ * C_ * 2);
    bf16_t* wv_bf = (bf16_t*)alloc((size_t)C_ * C_ * 2);
    bf16_t* wo_bf = (bf16_t*)alloc((size_t)C_ * C_ * 2);
    bf16_t* q_bf  = (bf16_t*)alloc((size_t)M_ * C_ * 2);
    bf16_t* k_bf  = (bf16_t*)alloc((size_t)M_ * C_ * 2);
    bf16_t* v_bf  = (bf16_t*)alloc((size_t)M_ * C_ * 2);
    bf16_t* xc_bf = (bf16_t*)alloc((size_t)M_ * C_ * 2);
    float*  partq = (float*)alloc((size_t)B_ * 32 * C_ * 4);
    float*  partk = (float*)alloc((size_t)B_ * 32 * C_ * 4);
    float*  invq  = (float*)alloc((size_t)B_ * C_ * 4);
    float*  invk  = (float*)alloc((size_t)B_ * C_ * 4);
    float*  partS = (float*)alloc((size_t)NCHUNK_ * B_ * H_ * 4096 * 4);
    bf16_t* p_bf  = (bf16_t*)alloc((size_t)B_ * H_ * 64 * 64 * 2);

    // 1) weights -> bf16
    cast_f32_to_bf16<<<dim3(C_ * C_ / 256), dim3(256), 0, stream>>>(Wq, wq_bf, C_ * C_);
    cast_f32_to_bf16<<<dim3(C_ * C_ / 256), dim3(256), 0, stream>>>(Wk, wk_bf, C_ * C_);
    cast_f32_to_bf16<<<dim3(C_ * C_ / 256), dim3(256), 0, stream>>>(Wv, wv_bf, C_ * C_);
    cast_f32_to_bf16<<<dim3(C_ * C_ / 256), dim3(256), 0, stream>>>(Wo, wo_bf, C_ * C_);

    // 2) projections (bf16 WMMA, fp32 accumulate)
    dim3 gproj(M_ / 64, C_ / 64);
    proj_gemm_kernel<<<gproj, dim3(256), 0, stream>>>(x1, wq_bf, q_bf);
    proj_gemm_kernel<<<gproj, dim3(256), 0, stream>>>(x2, wk_bf, k_bf);
    proj_gemm_kernel<<<gproj, dim3(256), 0, stream>>>(x2, wv_bf, v_bf);

    // 3) per-channel inverse L2 norms over the token axis
    colsumsq_partial<<<dim3(B_, 32), dim3(256), 0, stream>>>(q_bf, partq);
    colsumsq_partial<<<dim3(B_, 32), dim3(256), 0, stream>>>(k_bf, partk);
    finalize_inv_norm<<<dim3(B_), dim3(256), 0, stream>>>(partq, invq);
    finalize_inv_norm<<<dim3(B_), dim3(256), 0, stream>>>(partk, invk);

    // 4) channel-covariance logits (async-staged WMMA partials) + softmax
    attn_partial_kernel<<<dim3(NCHUNK_, B_ * H_), dim3(256), 0, stream>>>(
        q_bf, k_bf, partS);
    attn_softmax_finalize<<<dim3(B_ * H_), dim3(256), 0, stream>>>(
        partS, invq, invk, temp, p_bf);

    // 5) x_cross = attn @ v
    xcross_kernel<<<dim3(N_ / 64, B_ * H_), dim3(256), 0, stream>>>(v_bf, p_bf, xc_bf);

    // 6) output projection + bias
    out_gemm_kernel<<<gproj, dim3(256), 0, stream>>>(xc_bf, wo_bf, bo, out);
}
